// DMoELinear_35622458753618
// MI455X (gfx1250) — compile-verified, matrix operand order checked by reading, not run on hardware
//
#include <hip/hip_runtime.h>
#include <hip/hip_bf16.h>

typedef __attribute__((ext_vector_type(16))) __bf16 v16bf;
typedef __attribute__((ext_vector_type(8)))  __bf16 v8bf;
typedef __attribute__((ext_vector_type(8)))  float  v8f;

#define N_EXPERTS 8
#define IN_F      2048
#define OUT_F     2048
#define T_TOTAL   8192
#define M_TILE    128
#define N_TILE    128
#define K_CHUNK   64
#define NCHUNK    (IN_F / K_CHUNK)
#define LDA       72    /* padded row stride (bf16 elems) */
#define LDB       72

union V16 { v16bf v; v8bf h[2]; };

// ---------------------------------------------------------------------------
// Phase 0: zero the per-expert counters in workspace
// ---------------------------------------------------------------------------
__global__ void moe_init_counts(int* __restrict__ counts) {
    if (threadIdx.x < N_EXPERTS) counts[threadIdx.x] = 0;
}

// ---------------------------------------------------------------------------
// Phase 1: bucket token indices per expert (sorted grouped-GEMM dispatch)
// ---------------------------------------------------------------------------
__global__ void moe_build_lists(const int* __restrict__ ids,
                                int* __restrict__ counts,
                                int* __restrict__ lists) {
    int t = blockIdx.x * blockDim.x + threadIdx.x;
    if (t < T_TOTAL) {
        int e = ids[t] & (N_EXPERTS - 1);
        int pos = atomicAdd(&counts[e], 1);
        lists[e * T_TOTAL + pos] = t;
    }
}

// ---------------------------------------------------------------------------
// Phase 2: grouped GEMM. Block = 256 threads = 8 waves (wave32).
// Macro-tile 128 tokens x 128 cols; wave (mw,nw) in 4x2 grid owns a 32x64
// region = 2x4 fragments of v_wmma_f32_16x16x32_bf16.
// K streamed in 64-wide chunks through DOUBLE-BUFFERED LDS (one barrier per
// chunk). The chunk loop is FULLY UNROLLED: base pointers and per-thread
// address VGPRs are loop-invariant, the K-advance lives entirely in the
// 24-bit instruction immediate (max 7936 B < 2^23), so the steady state has
// zero address arithmetic.
// ---------------------------------------------------------------------------
__global__ __launch_bounds__(256)
void moe_gemm(const float* __restrict__ x,
              const float* __restrict__ w,
              const float* __restrict__ bias,
              const int*   __restrict__ counts,
              const int*   __restrict__ lists,
              __bf16*      __restrict__ out) {
    const int e   = blockIdx.z;
    const int cnt = counts[e];
    const int m0  = blockIdx.y * M_TILE;
    if (m0 >= cnt) return;                 // block-uniform: EXEC stays all-1s
    const int n0  = blockIdx.x * N_TILE;

    __shared__ __bf16 lA[2][M_TILE][LDA];
    __shared__ __bf16 lB[2][N_TILE][LDB];
    __shared__ int    toks[M_TILE];

    const int tid  = threadIdx.x;
    const int wave = tid >> 5;
    const int lane = tid & 31;
    const int lr   = lane & 15;
    const int half = lane >> 4;
    const int mw   = wave >> 1;            // 0..3 -> 32-row slice
    const int nw   = wave & 1;             // 0..1 -> 64-col slice

    // token gather list for this tile (pad rows clamp to a valid token)
    if (tid < M_TILE) {
        int r = m0 + tid;
        toks[tid] = lists[e * T_TOTAL + (r < cnt ? r : (cnt - 1))];
    }
    __syncthreads();

    // staging map: idx = tid + 256*j  ->  row/col = idx>>4, kq = idx&15.
    // Bases and offsets are invariant for the whole kernel; chunk advance is
    // a compile-time immediate after full unroll.
    const char* xb = reinterpret_cast<const char*>(x);
    const char* wb = reinterpret_cast<const char*>(w + (long)e * OUT_F * IN_F);
    int aOff[8], bOff[8];
    int srow[8], skk[8];
    #pragma unroll
    for (int j = 0; j < 8; ++j) {
        int idx = tid + 256 * j;
        srow[j] = idx >> 4;
        skk[j]  = (idx & 15) * 4;
        aOff[j] = toks[srow[j]] * (IN_F * 4) + (idx & 15) * 16;
        bOff[j] = (n0 + srow[j]) * (IN_F * 4) + (idx & 15) * 16;
    }

    v8f acc[2][4];
    #pragma unroll
    for (int mi = 0; mi < 2; ++mi)
        #pragma unroll
        for (int ni = 0; ni < 4; ++ni)
            acc[mi][ni] = (v8f){0.f,0.f,0.f,0.f,0.f,0.f,0.f,0.f};

    float4 aReg[8], bReg[8];

    // issue one chunk's global loads; `koff` is a compile-time byte offset
    // that folds into the instruction's 24-bit immediate
    auto load_chunk = [&](int koff) {
        #pragma unroll
        for (int j = 0; j < 8; ++j) {
            aReg[j] = *reinterpret_cast<const float4*>(xb + aOff[j] + koff);
            bReg[j] = *reinterpret_cast<const float4*>(wb + bOff[j] + koff);
        }
    };

    // spill staged registers (one chunk) into LDS buffer `buf`, fused cvt
    auto stage = [&](int buf) {
        #pragma unroll
        for (int j = 0; j < 8; ++j) {
            lA[buf][srow[j]][skk[j] + 0] = (__bf16)aReg[j].x;
            lA[buf][srow[j]][skk[j] + 1] = (__bf16)aReg[j].y;
            lA[buf][srow[j]][skk[j] + 2] = (__bf16)aReg[j].z;
            lA[buf][srow[j]][skk[j] + 3] = (__bf16)aReg[j].w;
            lB[buf][srow[j]][skk[j] + 0] = (__bf16)bReg[j].x;
            lB[buf][srow[j]][skk[j] + 1] = (__bf16)bReg[j].y;
            lB[buf][srow[j]][skk[j] + 2] = (__bf16)bReg[j].z;
            lB[buf][srow[j]][skk[j] + 3] = (__bf16)bReg[j].w;
        }
    };

    // ---- prologue: chunk 0 -> buf0, chunk 1 in flight in registers
    load_chunk(0);
    stage(0);
    load_chunk(K_CHUNK * 4);
    __syncthreads();

    #pragma unroll
    for (int i = 0; i < NCHUNK; ++i) {
        const int cur = i & 1;

        // stage chunk i+1 into the other buffer (overlaps this chunk's WMMAs)
        if (i + 1 < NCHUNK) stage(cur ^ 1);

        // launch global loads for chunk i+2 (pure immediate-offset loads)
        if (i + 2 < NCHUNK) load_chunk((i + 2) * (K_CHUNK * 4));

        // ---- 2 k-steps x (2x4) fragments = 16 WMMAs on buffer `cur`.
        // 16-bit operand lane layout (ISA 7.12.2): lanes 0-15 hold
        // K in {kb..kb+7, kb+16..kb+23}; lanes 16-31 are offset +8.
        #pragma unroll
        for (int s = 0; s < 2; ++s) {
            const int kb = s * 32 + half * 8;
            V16 af[2], bf[4];
            #pragma unroll
            for (int mi = 0; mi < 2; ++mi) {
                const __bf16* p = &lA[cur][mw * 32 + mi * 16 + lr][kb];
                af[mi].h[0] = *reinterpret_cast<const v8bf*>(p);
                af[mi].h[1] = *reinterpret_cast<const v8bf*>(p + 16);
            }
            #pragma unroll
            for (int ni = 0; ni < 4; ++ni) {
                const __bf16* p = &lB[cur][nw * 64 + ni * 16 + lr][kb];
                bf[ni].h[0] = *reinterpret_cast<const v8bf*>(p);
                bf[ni].h[1] = *reinterpret_cast<const v8bf*>(p + 16);
            }
            #pragma unroll
            for (int mi = 0; mi < 2; ++mi)
                #pragma unroll
                for (int ni = 0; ni < 4; ++ni)
                    acc[mi][ni] = __builtin_amdgcn_wmma_f32_16x16x32_bf16(
                        false, af[mi].v, false, bf[ni].v,
                        (short)0, acc[mi][ni], false, false);
        }
        __syncthreads();   // single barrier per chunk (double-buffered LDS)
    }

    // ---- epilogue: C/D layout -> VGPR r holds M=r (lanes 0-15) / M=r+8.
    char* ob = reinterpret_cast<char*>(out);
    const bool full = (cnt - m0) >= M_TILE;   // uniform across block
    if (full) {
        #pragma unroll
        for (int ni = 0; ni < 4; ++ni) {
            const int col = n0 + nw * 64 + ni * 16 + lr;
            const float bv = bias[e * OUT_F + col];
            #pragma unroll
            for (int mi = 0; mi < 2; ++mi)
                #pragma unroll
                for (int r = 0; r < 8; ++r) {
                    int m = mw * 32 + mi * 16 + r + half * 8;
                    int off = (toks[m] * OUT_F + col) * 2;
                    *reinterpret_cast<__bf16*>(ob + off) =
                        (__bf16)(acc[mi][ni][r] + bv);
                }
        }
    } else {
        #pragma unroll
        for (int ni = 0; ni < 4; ++ni) {
            const int col = n0 + nw * 64 + ni * 16 + lr;
            const float bv = bias[e * OUT_F + col];
            #pragma unroll
            for (int mi = 0; mi < 2; ++mi)
                #pragma unroll
                for (int r = 0; r < 8; ++r) {
                    int m = mw * 32 + mi * 16 + r + half * 8;
                    if (m0 + m < cnt) {
                        int off = (toks[m] * OUT_F + col) * 2;
                        *reinterpret_cast<__bf16*>(ob + off) =
                            (__bf16)(acc[mi][ni][r] + bv);
                    }
                }
        }
    }
}

// ---------------------------------------------------------------------------
extern "C" void kernel_launch(void* const* d_in, const int* in_sizes, int n_in,
                              void* d_out, int out_size, void* d_ws, size_t ws_size,
                              hipStream_t stream) {
    const float* x    = (const float*)d_in[0];   // [4,2048,2048] f32
    const int*   ids  = (const int*)  d_in[1];   // [4,2048] int
    const float* w    = (const float*)d_in[2];   // [8,2048,2048] f32
    const float* bias = (const float*)d_in[3];   // [8,2048] f32
    __bf16*      out  = (__bf16*)d_out;          // [8192,2048] bf16

    int* counts = (int*)d_ws;                          // 8 ints
    int* lists  = (int*)((char*)d_ws + 256);           // 8 x 8192 ints

    moe_init_counts<<<1, 32, 0, stream>>>(counts);
    moe_build_lists<<<(T_TOTAL + 255) / 256, 256, 0, stream>>>(ids, counts, lists);

    dim3 grid(OUT_F / N_TILE, (T_TOTAL + M_TILE - 1) / M_TILE, N_EXPERTS);
    moe_gemm<<<grid, 256, 0, stream>>>(x, w, bias, counts, lists, out);
}
